// GCN_82978768159016
// MI455X (gfx1250) — matrix operand order repaired and checked by reading, not compiled, not run on hardware
//
#include <hip/hip_runtime.h>
#include <hip/hip_bf16.h>

typedef __attribute__((ext_vector_type(16))) _Float16 v16h;
typedef __attribute__((ext_vector_type(8)))  float    v8f;

#define D_FEAT   128
#define DE_FEAT  16
#define KMSG     160   // 144 padded to 5 K-blocks of 32
#define H1_FEAT  256
#define P_FEAT   128
#define NG       64

// ---------------------------------------------------------------------------
// A-fragment loader for V_WMMA_F32_16X16X32_F16.
// ISA 7.12.2, 16-bit A 16x32: lanes 0-15: V0..3 -> K=0..7, V4..7 -> K=16..23;
// lanes 16-31: +8. Each dword holds two consecutive halfs (low 16 = even K).
__device__ __forceinline__ v16h load_a_frag(const _Float16* arow, int khalf) {
    union { v16h v; unsigned int u[8]; } av;
#pragma unroll
    for (int r = 0; r < 8; ++r) {
        int koff = ((r < 4) ? (2 * r) : (16 + 2 * (r - 4))) + khalf;
        av.u[r] = *(const unsigned int*)(arow + koff);
    }
    return av.v;
}

// ---------------------------------------------------------------------------
// Degree count + per-node sum of incoming edge attrs (for self-loop fill)
__global__ void edge_accum(const int* __restrict__ col,
                           const float* __restrict__ edge_attr,
                           float* __restrict__ in_cnt,
                           float* __restrict__ loop_attr, int E) {
    int e = blockIdx.x * 256 + threadIdx.x;
    if (e >= E) return;
    int c = col[e];
    atomicAdd(&in_cnt[c], 1.0f);
    const float4* a = (const float4*)(edge_attr + (long)e * DE_FEAT);
    float* d = loop_attr + (long)c * DE_FEAT;
#pragma unroll
    for (int q = 0; q < 4; ++q) {
        float4 v = a[q];
        atomicAdd(d + 4 * q + 0, v.x);
        atomicAdd(d + 4 * q + 1, v.y);
        atomicAdd(d + 4 * q + 2, v.z);
        atomicAdd(d + 4 * q + 3, v.w);
    }
}

// loop_attr /= max(cnt,1); norm = rsqrt(deg incl. self loop); graph counts
__global__ void node_prep(const float* __restrict__ in_cnt,
                          float* __restrict__ loop_attr,
                          float* __restrict__ norm,
                          const int* __restrict__ batch,
                          float* __restrict__ gcnt, int N) {
    int i = blockIdx.x * 256 + threadIdx.x;
    if (i >= N) return;
    float c = in_cnt[i];
    norm[i] = rsqrtf(c + 1.0f);              // deg includes self loop, >= 1
    float inv = 1.0f / fmaxf(c, 1.0f);
#pragma unroll
    for (int j = 0; j < DE_FEAT; ++j) loop_attr[(long)i * DE_FEAT + j] *= inv;
    atomicAdd(&gcnt[batch[i]], 1.0f);
}

__global__ void x_to_f16(const float* __restrict__ x, _Float16* __restrict__ xh, long n) {
    long i = (long)blockIdx.x * 256 + threadIdx.x;
    if (i < n) xh[i] = (_Float16)x[i];
}

// Fused edge list: real edges then self loops, with symmetric norm weight
__global__ void build_edges(const int* __restrict__ row, const int* __restrict__ col,
                            const float* __restrict__ norm,
                            int* __restrict__ row32, int* __restrict__ col32,
                            float* __restrict__ ew, int E, int N) {
    int e = blockIdx.x * 256 + threadIdx.x;
    int ET = E + N;
    if (e >= ET) return;
    int r = (e < E) ? row[e] : (e - E);
    int c = (e < E) ? col[e] : (e - E);
    row32[e] = r;
    col32[e] = c;
    ew[e] = norm[r] * norm[c];
}

// ---------------------------------------------------------------------------
// Pre-pack weights into per-lane B-fragment order: lane L (of the wave that
// owns n-tile `ntile`, K-block `kb`) holds K = kb*32 + (e<8 ? 8*(L>=16)+e : 16+8*(L>=16)+e-8),
// N = ntile*16 + (L&15), as 16 contiguous f16 (one 32B load in the GEMM).
__global__ void prepack_wmsg(const float* __restrict__ W, _Float16* __restrict__ P) {
    int t = blockIdx.x * 256 + threadIdx.x;           // 8 ntiles * 5 kb * 32 * 16
    if (t >= 8 * 5 * 32 * 16) return;
    int e = t & 15, lane = (t >> 4) & 31, kb = (t >> 9) % 5, ntile = (t >> 9) / 5;
    int K = kb * 32 + ((e < 8) ? 0 : 16) + ((lane >= 16) ? 8 : 0) + (e & 7);
    int n = ntile * 16 + (lane & 15);
    P[t] = (_Float16)((K < D_FEAT + DE_FEAT) ? W[(long)K * D_FEAT + n] : 0.0f);
}

__global__ void prepack_wapply(const float* __restrict__ W, _Float16* __restrict__ P) {
    int t = blockIdx.x * 256 + threadIdx.x;           // 16 ntiles * 8 kb * 32 * 16
    if (t >= 16 * 8 * 32 * 16) return;
    int e = t & 15, lane = (t >> 4) & 31, kb = (t >> 9) % 8, ntile = (t >> 9) / 8;
    int K = kb * 32 + ((e < 8) ? 0 : 16) + ((lane >= 16) ? 8 : 0) + (e & 7);
    int n = ntile * 16 + (lane & 15);
    P[t] = (_Float16)W[(long)K * H1_FEAT + n];
}

// ---------------------------------------------------------------------------
// Message GEMM + scatter: per block, 16 edges x 128 features.
// msg = relu([x[row] | attr] @ W_msg + b_msg) * w_e, atomically added to aggr[col].
__global__ void __launch_bounds__(256)
msg_wmma(const _Float16* __restrict__ xh,
         const float* __restrict__ edge_attr, const float* __restrict__ loop_attr,
         const int* __restrict__ row32, const int* __restrict__ col32,
         const float* __restrict__ ew,
         const _Float16* __restrict__ packW, const float* __restrict__ b_msg,
         float* __restrict__ aggr, int ET, int E) {
    __shared__ __align__(16) _Float16 lA[16 * KMSG];
    __shared__ int s_row[16], s_col[16];
    __shared__ float s_w[16];

    int tid = threadIdx.x;
    int eb  = blockIdx.x * 16;
    if (tid < 16) {
        int e = eb + tid;
        if (e < ET) { s_row[tid] = row32[e]; s_col[tid] = col32[e]; s_w[tid] = ew[e]; }
        else        { s_row[tid] = 0;        s_col[tid] = 0;        s_w[tid] = 0.0f; }
    }
    __syncthreads();

    // Stage A tile: 16 rows x [ x_f16(128) | attr(16) | pad(16) ]
    for (int t = tid; t < 16 * KMSG; t += 256) {
        int m = t / KMSG, k = t % KMSG;
        int e = eb + m;
        _Float16 v = (_Float16)0.0f;
        if (e < ET) {
            if (k < D_FEAT) {
                v = xh[(long)s_row[m] * D_FEAT + k];
            } else if (k < D_FEAT + DE_FEAT) {
                int j = k - D_FEAT;
                float a = (e < E) ? edge_attr[(long)e * DE_FEAT + j]
                                  : loop_attr[(long)(e - E) * DE_FEAT + j];
                v = (_Float16)a;
            }
        }
        lA[t] = v;
    }
    __syncthreads();

    int wave = tid >> 5, lane = tid & 31;
    int kh   = (lane >= 16) ? 8 : 0;     // A K-half select / C M-half select
    v8f acc = {};
#pragma unroll
    for (int kb = 0; kb < 5; ++kb) {
        v16h a = load_a_frag(&lA[(lane & 15) * KMSG + kb * 32], kh);
        v16h b = *(const v16h*)(packW + (((long)(wave * 5 + kb) * 32 + lane) << 4));
        acc = __builtin_amdgcn_wmma_f32_16x16x32_f16(false, a, false, b,
                                                     (short)0, acc, false, false);
    }
    // C/D layout: VGPR r -> M = r + 8*(lane>=16), N = wave*16 + (lane&15)
    int n = wave * 16 + (lane & 15);
    float bn = b_msg[n];
#pragma unroll
    for (int r = 0; r < 8; ++r) {
        int mm = r + kh;
        float val = fmaxf(acc[r] + bn, 0.0f) * s_w[mm];
        atomicAdd(&aggr[(long)s_col[mm] * D_FEAT + n], val);
    }
}

// ---------------------------------------------------------------------------
// Apply GEMM + fused mean-pool accumulation: h = relu([x|aggr]@W_apply + b),
// g_accum[batch[i]] += h[i].
__global__ void __launch_bounds__(256)
apply_wmma(const _Float16* __restrict__ xh, const float* __restrict__ aggr,
           const _Float16* __restrict__ packW, const float* __restrict__ b_apply,
           const int* __restrict__ batch, float* __restrict__ g_accum, int N) {
    __shared__ __align__(16) _Float16 lA[16 * 256];
    __shared__ int s_batch[16];

    int tid = threadIdx.x;
    int nb  = blockIdx.x * 16;
    if (tid < 16) {
        int node = nb + tid;
        s_batch[tid] = (node < N) ? batch[node] : -1;
    }
    __syncthreads();

    for (int t = tid; t < 16 * 256; t += 256) {
        int m = t >> 8, k = t & 255;
        int node = nb + m;
        _Float16 v = (_Float16)0.0f;
        if (node < N)
            v = (k < D_FEAT) ? xh[(long)node * D_FEAT + k]
                             : (_Float16)aggr[(long)node * D_FEAT + (k - D_FEAT)];
        lA[t] = v;
    }
    __syncthreads();

    int wave = tid >> 5, lane = tid & 31;
    int kh   = (lane >= 16) ? 8 : 0;
#pragma unroll
    for (int half = 0; half < 2; ++half) {
        int ntile = wave + half * 8;          // 16 n-tiles cover 256 features
        v8f acc = {};
#pragma unroll
        for (int kb = 0; kb < 8; ++kb) {
            v16h a = load_a_frag(&lA[(lane & 15) * 256 + kb * 32], kh);
            v16h b = *(const v16h*)(packW + (((long)(ntile * 8 + kb) * 32 + lane) << 4));
            acc = __builtin_amdgcn_wmma_f32_16x16x32_f16(false, a, false, b,
                                                         (short)0, acc, false, false);
        }
        int n = ntile * 16 + (lane & 15);
        float bn = b_apply[n];
#pragma unroll
        for (int r = 0; r < 8; ++r) {
            int bg = s_batch[r + kh];
            if (bg >= 0) {
                float val = fmaxf(acc[r] + bn, 0.0f);
                atomicAdd(&g_accum[(long)bg * H1_FEAT + n], val);
            }
        }
    }
}

// ---------------------------------------------------------------------------
// Head: mean pool divide + BatchNorm (batch stats) + relu(g@W1+b1)@W2+b2
__global__ void __launch_bounds__(256)
head_kernel(const float* __restrict__ g_accum, const float* __restrict__ gcnt,
            const float* __restrict__ gamma, const float* __restrict__ beta,
            const float* __restrict__ W1, const float* __restrict__ b1,
            const float* __restrict__ W2, const float* __restrict__ b2,
            float* __restrict__ g_norm, float* __restrict__ out) {
    __shared__ float t1[NG * P_FEAT];
    int tid = threadIdx.x;

    {   // one feature per thread, BN over the 64 graphs
        int f = tid;
        float gv[NG];
        float mu = 0.0f;
#pragma unroll
        for (int b = 0; b < NG; ++b) {
            gv[b] = g_accum[b * H1_FEAT + f] / fmaxf(gcnt[b], 1.0f);
            mu += gv[b];
        }
        mu *= (1.0f / NG);
        float var = 0.0f;
#pragma unroll
        for (int b = 0; b < NG; ++b) { float d = gv[b] - mu; var += d * d; }
        var *= (1.0f / NG);
        float s = rsqrtf(var + 1e-5f) * gamma[f];
        float bt = beta[f];
#pragma unroll
        for (int b = 0; b < NG; ++b) g_norm[b * H1_FEAT + f] = (gv[b] - mu) * s + bt;
    }
    __syncthreads();

    for (int t = tid; t < NG * P_FEAT; t += 256) {
        int b = t >> 7, p = t & (P_FEAT - 1);
        float acc = b1[p];
        for (int f = 0; f < H1_FEAT; ++f) acc += g_norm[b * H1_FEAT + f] * W1[f * P_FEAT + p];
        t1[t] = fmaxf(acc, 0.0f);
    }
    __syncthreads();

    if (tid < NG) {
        float acc = b2[0];
        for (int p = 0; p < P_FEAT; ++p) acc += t1[tid * P_FEAT + p] * W2[p];
        out[tid] = acc;
    }
}

// ---------------------------------------------------------------------------
extern "C" void kernel_launch(void* const* d_in, const int* in_sizes, int n_in,
                              void* d_out, int out_size, void* d_ws, size_t ws_size,
                              hipStream_t stream) {
    const float* x         = (const float*)d_in[0];
    const int*   edge_idx  = (const int*)d_in[1];
    const float* edge_attr = (const float*)d_in[2];
    const int*   batch     = (const int*)d_in[3];
    const float* W_msg     = (const float*)d_in[4];
    const float* b_msg     = (const float*)d_in[5];
    const float* W_apply   = (const float*)d_in[6];
    const float* b_apply   = (const float*)d_in[7];
    const float* bn_gamma  = (const float*)d_in[8];
    const float* bn_beta   = (const float*)d_in[9];
    const float* W1        = (const float*)d_in[10];
    const float* b1        = (const float*)d_in[11];
    const float* W2        = (const float*)d_in[12];
    const float* b2        = (const float*)d_in[13];
    float* out = (float*)d_out;

    const int N  = in_sizes[3];
    const int E  = in_sizes[1] / 2;
    const int ET = E + N;
    const int* row = edge_idx;
    const int* col = edge_idx + E;

    // Workspace carve-up (256B aligned chunks)
    size_t off = 0;
    auto alloc = [&](size_t bytes) -> void* {
        off = (off + 255) & ~(size_t)255;
        void* p = (char*)d_ws + off;
        off += bytes;
        return p;
    };
    float*    in_cnt    = (float*)alloc((size_t)N * 4);
    float*    loop_attr = (float*)alloc((size_t)N * DE_FEAT * 4);
    float*    norm      = (float*)alloc((size_t)N * 4);
    _Float16* xh        = (_Float16*)alloc((size_t)N * D_FEAT * 2);
    int*      row32     = (int*)alloc((size_t)ET * 4);
    int*      col32     = (int*)alloc((size_t)ET * 4);
    float*    ew        = (float*)alloc((size_t)ET * 4);
    float*    aggr      = (float*)alloc((size_t)N * D_FEAT * 4);
    _Float16* pWmsg     = (_Float16*)alloc((size_t)8 * 5 * 32 * 16 * 2);
    _Float16* pWapply   = (_Float16*)alloc((size_t)16 * 8 * 32 * 16 * 2);
    float*    g_accum   = (float*)alloc((size_t)NG * H1_FEAT * 4);
    float*    gcnt      = (float*)alloc((size_t)NG * 4);
    float*    g_norm    = (float*)alloc((size_t)NG * H1_FEAT * 4);
    (void)ws_size; (void)n_in; (void)out_size;

    // Zero the accumulators (graph-capture-safe memset nodes)
    hipMemsetAsync(in_cnt,    0, (size_t)N * 4, stream);
    hipMemsetAsync(loop_attr, 0, (size_t)N * DE_FEAT * 4, stream);
    hipMemsetAsync(aggr,      0, (size_t)N * D_FEAT * 4, stream);
    hipMemsetAsync(g_accum,   0, (size_t)NG * H1_FEAT * 4, stream);
    hipMemsetAsync(gcnt,      0, (size_t)NG * 4, stream);

    edge_accum<<<(E + 255) / 256, 256, 0, stream>>>(col, edge_attr, in_cnt, loop_attr, E);
    node_prep<<<(N + 255) / 256, 256, 0, stream>>>(in_cnt, loop_attr, norm, batch, gcnt, N);
    {
        long nx = (long)N * D_FEAT;
        x_to_f16<<<(unsigned)((nx + 255) / 256), 256, 0, stream>>>(x, xh, nx);
    }
    build_edges<<<(ET + 255) / 256, 256, 0, stream>>>(row, col, norm, row32, col32, ew, E, N);
    prepack_wmsg<<<(8 * 5 * 32 * 16 + 255) / 256, 256, 0, stream>>>(W_msg, pWmsg);
    prepack_wapply<<<(16 * 8 * 32 * 16 + 255) / 256, 256, 0, stream>>>(W_apply, pWapply);

    msg_wmma<<<(ET + 15) / 16, 256, 0, stream>>>(xh, edge_attr, loop_attr, row32, col32, ew,
                                                 pWmsg, b_msg, aggr, ET, E);
    apply_wmma<<<(N + 15) / 16, 256, 0, stream>>>(xh, aggr, pWapply, b_apply, batch, g_accum, N);
    head_kernel<<<1, 256, 0, stream>>>(g_accum, gcnt, bn_gamma, bn_beta, W1, b1, W2, b2,
                                       g_norm, out);
}
